// VRNN_15582141350169
// MI455X (gfx1250) — matrix operand order repaired and checked by reading, not compiled, not run on hardware
//
#include <hip/hip_runtime.h>
#include <cstdint>
#include <cstddef>

// ---------------------------------------------------------------------------
// VRNN forward for MI455X (gfx1250): bf16 WMMA GEMMs, async global->LDS
// staging (ASYNCcnt), double-buffered tiles, fused epilogues.
// ---------------------------------------------------------------------------

typedef __attribute__((ext_vector_type(16))) __bf16          v16bf;
typedef __attribute__((ext_vector_type(8)))  float           v8f;
typedef __attribute__((ext_vector_type(16))) unsigned short  v16us;
typedef __attribute__((ext_vector_type(8)))  unsigned short  v8us;

#define T_STEPS 160
#define BATCH   512
#define XD      88
#define XDP     96      // X padded to K multiple of 32 (and N multiple of 16)
#define HD      512
#define ZD      128

__device__ __forceinline__ unsigned short f2bf(float f) {
  unsigned int u = __float_as_uint(f);
  u += 0x7FFFu + ((u >> 16) & 1u);          // round-to-nearest-even
  return (unsigned short)(u >> 16);
}
__device__ __forceinline__ float sigmoidf_(float x) { return 1.0f / (1.0f + __expf(-x)); }
__device__ __forceinline__ float softplusf_(float x) {
  return fmaxf(x, 0.0f) + log1pf(__expf(-fabsf(x)));
}

template <int ACT>
__device__ __forceinline__ float apply_act(float v) {
  if (ACT == 1) return fmaxf(v, 0.0f);        // relu
  if (ACT == 2) return sigmoidf_(v);          // sigmoid
  if (ACT == 3) return softplusf_(v);         // softplus
  if (ACT == 4) return softplusf_(v) + 1.0f;  // softplus + 1 (prior_std)
  return v;
}

// ---- CDNA5 async global->LDS copy (ASYNCcnt) ------------------------------
// GLOBAL_LOAD_ASYNC_TO_LDS_B128: VDST = LDS byte address (VGPR),
// VADDR = 64-bit global address. Tracked by ASYNCcnt (ISA 08 §4).
__device__ __forceinline__ void async_ld16(unsigned short* lds, const unsigned short* g) {
  __attribute__((address_space(3))) unsigned short* l =
      (__attribute__((address_space(3))) unsigned short*)lds;
  const __attribute__((address_space(1))) unsigned short* p =
      (const __attribute__((address_space(1))) unsigned short*)g;
  asm volatile("global_load_async_to_lds_b128 %0, %1, off"
               :: "v"(l), "v"(p) : "memory");
}
__device__ __forceinline__ void wait_async0() {
  asm volatile("s_wait_asynccnt 0" ::: "memory");
}

// ---------------------------------------------------------------------------
// Tiled bf16 GEMM:  out[M,N] = act(A[M,K] @ W[N,K]^T + bias)
//   block tile 64(M) x 128(N), BK=32; 8 waves, each wave: 16(M) x 64(N)
//   -> 4x v_wmma_f32_16x16x32_bf16 per K-step per wave, issued back-to-back
//      behind a single s_wait_dscnt, with the A-reuse hint on the first three.
//   Double-buffered LDS tiles filled with global_load_async_to_lds_b128.
// Requires: M % 64 == 0, K % 32 == 0. N handled with guards (zero-fill LDS).
// outF: f32 result (row stride ldo), outH: bf16 result (row stride N).
// ---------------------------------------------------------------------------
template <int ACT>
__global__ __launch_bounds__(256)
void gemm_bf16_wmma(const unsigned short* __restrict__ A,
                    const unsigned short* __restrict__ W,
                    const float* __restrict__ bias,
                    float* __restrict__ outF, int ldo,
                    unsigned short* __restrict__ outH,
                    int M, int N, int K, int Nout) {
  __shared__ __align__(64) unsigned short As[2][64 * 32];
  __shared__ __align__(64) unsigned short Bs[2][128 * 32];

  const int tid  = threadIdx.x;
  const int lane = tid & 31;
  const int wave = tid >> 5;
  const int wm = wave >> 1;                 // 0..3 : M sub-tile
  const int wn = wave & 1;                  // 0..1 : N sub-tile
  const int mBlock = blockIdx.y * 64;
  const int nBlock = blockIdx.x * 128;

  v8f acc[4] = {};

  // cooperative tile-load assignments
  const int ar = tid >> 2;                  // 0..63 : A row, 8 bf16 (16B) each
  const int ac = (tid & 3) * 8;
  const int bn = tid >> 1;                  // 0..127 : weight row n, 16 bf16 (32B)
  const int bc = (tid & 1) * 16;
  const unsigned short* aGlob = A + (size_t)(mBlock + ar) * K + ac;
  const int bnG = nBlock + bn;
  const unsigned short* bGlob = W + (size_t)bnG * K + bc;
  const bool bValid = (bnG < N);

  // zero-fill out-of-range weight rows once (contents never change after)
  if (!bValid) {
    v16us z = {};
    *(v16us*)&Bs[0][bn * 32 + bc] = z;
    *(v16us*)&Bs[1][bn * 32 + bc] = z;
  }

  // prologue: stage K-tile 0 into buffer 0 (async DMA to LDS)
  async_ld16(&As[0][ar * 32 + ac], aGlob);
  if (bValid) {
    async_ld16(&Bs[0][bn * 32 + bc], bGlob);
    async_ld16(&Bs[0][bn * 32 + bc + 8], bGlob + 8);
    if (32 < K) __builtin_prefetch(bGlob + 32, 0, 1);   // global_prefetch_b8
  }

  int cur = 0;
  for (int k0 = 0; k0 < K; k0 += 32) {
    wait_async0();            // my async tile landed in LDS
    __syncthreads();          // everyone's tile landed; prev buffer free

    const int nxt = cur ^ 1;
    if (k0 + 32 < K) {        // stage next K-tile into the other buffer
      async_ld16(&As[nxt][ar * 32 + ac], aGlob + k0 + 32);
      if (bValid) {
        async_ld16(&Bs[nxt][bn * 32 + bc], bGlob + k0 + 32);
        async_ld16(&Bs[nxt][bn * 32 + bc + 8], bGlob + k0 + 32 + 8);
      }
    }

    // ---- A fragment (16x32 MxK): lane m = l%16; l/16 picks K-halves ----
    const int lm = lane & 15;
    const int lh = lane >> 4;
    const unsigned short* ap = &As[cur][(wm * 16 + lm) * 32 + lh * 16];
    v8us alo = *(const v8us*)ap;            // K = lh*16 .. +7
    v8us ahi = *(const v8us*)(ap + 8);      // K = lh*16 + 8 .. +15
    v16us au;
#pragma unroll
    for (int i = 0; i < 8; ++i) { au[i] = alo[i]; au[8 + i] = ahi[i]; }
    v16bf af = __builtin_bit_cast(v16bf, au);

    // ---- 4 B fragments (32x16 KxN): lane n = l%16; l/16 picks K 0-15/16-31
    const unsigned short* bbase = &Bs[cur][(wn * 64 + lm) * 32 + lh * 16];
    v16us bu0 = *(const v16us*)(bbase + 0 * 16 * 32);
    v16us bu1 = *(const v16us*)(bbase + 1 * 16 * 32);
    v16us bu2 = *(const v16us*)(bbase + 2 * 16 * 32);
    v16us bu3 = *(const v16us*)(bbase + 3 * 16 * 32);
    // Scheduling fence: force all fragment loads to complete here so the four
    // WMMAs below issue back-to-back behind a single s_wait_dscnt.
    asm volatile("" : "+v"(bu0), "+v"(bu1), "+v"(bu2), "+v"(bu3));

    acc[0] = __builtin_amdgcn_wmma_f32_16x16x32_bf16(
        false, af, false, __builtin_bit_cast(v16bf, bu0), (short)0, acc[0],
        /*reuse_a=*/true, false);
    acc[1] = __builtin_amdgcn_wmma_f32_16x16x32_bf16(
        false, af, false, __builtin_bit_cast(v16bf, bu1), (short)0, acc[1],
        /*reuse_a=*/true, false);
    acc[2] = __builtin_amdgcn_wmma_f32_16x16x32_bf16(
        false, af, false, __builtin_bit_cast(v16bf, bu2), (short)0, acc[2],
        /*reuse_a=*/true, false);
    acc[3] = __builtin_amdgcn_wmma_f32_16x16x32_bf16(
        false, af, false, __builtin_bit_cast(v16bf, bu3), (short)0, acc[3],
        false, false);
    cur = nxt;
  }

  // ---- epilogue: C/D layout -> row = lh*8 + e, col = lm ----
  const int lm = lane & 15;
  const int lh = lane >> 4;
#pragma unroll
  for (int j = 0; j < 4; ++j) {
    int col = nBlock + wn * 64 + j * 16 + lm;
    if (col >= Nout) continue;
    float b = bias ? bias[col] : 0.0f;
#pragma unroll
    for (int e = 0; e < 8; ++e) {
      int row = mBlock + wm * 16 + lh * 8 + e;
      float v = apply_act<ACT>(acc[j][e] + b);
      if (outF) outF[(size_t)row * ldo + col] = v;
      if (outH) outH[(size_t)row * N + col] = f2bf(v);
    }
  }
}

// ---------------------------------------------------------------------------
// Elementwise / setup kernels
// ---------------------------------------------------------------------------
__global__ void cvt_pad_kernel(unsigned short* __restrict__ dst,
                               const float* __restrict__ src,
                               int srcR, int srcC, int dstR, int dstC) {
  int i = blockIdx.x * 256 + threadIdx.x;
  if (i >= dstR * dstC) return;
  int r = i / dstC, c = i - r * dstC;
  float v = (r < srcR && c < srcC) ? src[r * srcC + c] : 0.0f;
  dst[i] = f2bf(v);
}

__global__ void lstm_pack_kernel(unsigned short* __restrict__ dst,
                                 const float* __restrict__ wih,
                                 const float* __restrict__ whh) {
  int i = blockIdx.x * 256 + threadIdx.x;
  if (i >= 2048 * 1536) return;
  int r = i / 1536, c = i - r * 1536;
  float v = (c < 1024) ? wih[r * 1024 + c] : whh[r * 512 + (c - 1024)];
  dst[i] = f2bf(v);
}

__global__ void init_state_kernel(unsigned short* __restrict__ h,
                                  float* __restrict__ c,
                                  float* __restrict__ out01) {
  int i = blockIdx.x * 256 + threadIdx.x;
  if (i < BATCH * HD) { h[i] = 0; c[i] = 0.0f; }
  if (i < 2) out01[i] = 0.0f;
}

// catA = [phi_x | h]; catC[:, :512]=phi_x, catC[:, 1024:]=h
__global__ void cat_start_kernel(unsigned short* __restrict__ catA,
                                 unsigned short* __restrict__ catC,
                                 const unsigned short* __restrict__ phix,
                                 const unsigned short* __restrict__ h) {
  int i = blockIdx.x * 256 + threadIdx.x;
  if (i >= BATCH * 1024) return;
  int r = i >> 10, c = i & 1023;
  unsigned short v = (c < 512) ? phix[(r << 9) + c] : h[(r << 9) + (c - 512)];
  catA[i] = v;
  catC[r * 1536 + ((c < 512) ? c : (c + 512))] = v;
}

// catB = [phi_z | h]; catC[:, 512:1024] = phi_z
__global__ void cat_mid_kernel(unsigned short* __restrict__ catB,
                               unsigned short* __restrict__ catC,
                               const unsigned short* __restrict__ phiz,
                               const unsigned short* __restrict__ h) {
  int i = blockIdx.x * 256 + threadIdx.x;
  if (i >= BATCH * 1024) return;
  int r = i >> 10, c = i & 1023;
  unsigned short v = (c < 512) ? phiz[(r << 9) + c] : h[(r << 9) + (c - 512)];
  catB[i] = v;
  if (c < 512) catC[r * 1536 + 512 + c] = v;
}

// z = eps * enc_std + enc_mean  (bf16 out, feeds phi_z GEMM)
__global__ void z_kernel(unsigned short* __restrict__ zbf,
                         const float* __restrict__ eps_t,
                         const float* __restrict__ em,
                         const float* __restrict__ es) {
  int i = blockIdx.x * 256 + threadIdx.x;
  if (i >= BATCH * ZD) return;
  zbf[i] = f2bf(eps_t[i] * es[i] + em[i]);
}

__global__ void kld_kernel(float* __restrict__ kld,
                           const float* __restrict__ em, const float* __restrict__ es,
                           const float* __restrict__ pm, const float* __restrict__ ps,
                           const float* __restrict__ mask_t) {
  __shared__ float red[128];
  int r = blockIdx.x, c = threadIdx.x;
  int i = r * ZD + c;
  float e = es[i], p = ps[i], dm = em[i] - pm[i];
  float el = 2.0f * __logf(p) - 2.0f * __logf(e) + (e * e + dm * dm) / (p * p) - 1.0f;
  red[c] = el;
  __syncthreads();
  for (int s = 64; s > 0; s >>= 1) { if (c < s) red[c] += red[c + s]; __syncthreads(); }
  if (c == 0) atomicAdd(kld, 0.5f * red[0] * mask_t[r]);
}

__global__ void nll_kernel(float* __restrict__ nll,
                           const float* __restrict__ x_t,
                           const float* __restrict__ dm,
                           const float* __restrict__ mask_t) {
  __shared__ float red[128];
  int r = blockIdx.x, c = threadIdx.x;
  float el = 0.0f;
  if (c < XD) {
    float xm = x_t[r * XD + c];
    float d  = dm[r * XD + c];
    el = -(xm * __logf(d) + (1.0f - xm) * __logf(1.0f - d));
  }
  red[c] = el;
  __syncthreads();
  for (int s = 64; s > 0; s >>= 1) { if (c < s) red[c] += red[c + s]; __syncthreads(); }
  if (c == 0) atomicAdd(nll, red[0] * mask_t[r]);
}

__global__ void lstm_kernel(unsigned short* __restrict__ h,
                            float* __restrict__ c,
                            const float* __restrict__ gates) {
  int i = blockIdx.x * 256 + threadIdx.x;
  if (i >= BATCH * HD) return;
  int r = i >> 9, cc = i & 511;
  const float* g = gates + (size_t)r * 2048;
  float ig = sigmoidf_(g[cc]);
  float fg = sigmoidf_(g[512 + cc]);
  float gg = tanhf(g[1024 + cc]);
  float og = sigmoidf_(g[1536 + cc]);
  float cn = fg * c[i] + ig * gg;
  c[i] = cn;
  h[i] = f2bf(og * tanhf(cn));
}

// ---------------------------------------------------------------------------
// Host orchestration
// ---------------------------------------------------------------------------
extern "C" void kernel_launch(void* const* d_in, const int* in_sizes, int n_in,
                              void* d_out, int out_size, void* d_ws, size_t ws_size,
                              hipStream_t stream) {
  (void)in_sizes; (void)n_in; (void)out_size; (void)ws_size;

  const float* x    = (const float*)d_in[0];   // [160,512,88]
  const float* mask = (const float*)d_in[1];   // [160,512]
  const float* eps  = (const float*)d_in[2];   // [160,512,128]
  const float* P[30];
  for (int i = 0; i < 30; ++i) P[i] = (const float*)d_in[3 + i];
  enum { PHI_X_W1 = 0, PHI_X_B1, PHI_X_W2, PHI_X_B2, PHI_Z_W, PHI_Z_B,
         ENC_W1, ENC_B1, ENC_W2, ENC_B2, ENC_MEAN_W, ENC_MEAN_B,
         ENC_STD_W, ENC_STD_B, PRIOR_W, PRIOR_B, PRIOR_MEAN_W, PRIOR_MEAN_B,
         PRIOR_STD_W, PRIOR_STD_B, DEC_W1, DEC_B1, DEC_W2, DEC_B2,
         DEC_STD_W, DEC_STD_B, DEC_MEAN_W, DEC_MEAN_B, LSTM_W_IH, LSTM_W_HH };

  float* out  = (float*)d_out;
  float* kld  = out;
  float* nll  = out + 1;
  float* encM = out + 2;
  float* encS = encM + (size_t)T_STEPS * BATCH * ZD;
  float* decM = encS + (size_t)T_STEPS * BATCH * ZD;
  float* decS = decM + (size_t)T_STEPS * BATCH * XD;

  // ---- carve workspace ----
  char* wsp = (char*)d_ws;
  auto carve = [&](size_t bytes) -> void* {
    void* r = (void*)wsp;
    wsp += (bytes + 255) & ~(size_t)255;
    return r;
  };
  auto cu = [&](size_t n) { return (unsigned short*)carve(n * 2); };
  auto cf = [&](size_t n) { return (float*)carve(n * 4); };

  unsigned short* w_phix1 = cu(512 * XDP);
  unsigned short* w_phix2 = cu(512 * 512);
  unsigned short* w_phiz  = cu(512 * 128);
  unsigned short* w_enc1  = cu(512 * 1024);
  unsigned short* w_enc2  = cu(512 * 512);
  unsigned short* w_encm  = cu(128 * 512);
  unsigned short* w_encs  = cu(128 * 512);
  unsigned short* w_prior = cu(512 * 512);
  unsigned short* w_priorm = cu(128 * 512);
  unsigned short* w_priors = cu(128 * 512);
  unsigned short* w_dec1  = cu(512 * 1024);
  unsigned short* w_dec2  = cu(512 * 512);
  unsigned short* w_decm  = cu(96 * 512);
  unsigned short* w_decs  = cu(96 * 512);
  unsigned short* w_lstm  = cu(2048 * 1536);

  unsigned short* xpad     = cu((size_t)8192 * XDP);          // chunk scratch
  unsigned short* h1chunk  = cu((size_t)8192 * 512);          // chunk scratch
  unsigned short* phix_all = cu((size_t)T_STEPS * BATCH * HD);// 160*512*512 bf16

  unsigned short* h_buf = cu(BATCH * HD);
  float*          c_buf = cf(BATCH * HD);
  unsigned short* catA  = cu(BATCH * 1024);
  unsigned short* catB  = cu(BATCH * 1024);
  unsigned short* catC  = cu(BATCH * 1536);
  unsigned short* ench1 = cu(BATCH * HD);
  unsigned short* ench  = cu(BATCH * HD);
  unsigned short* priorh = cu(BATCH * HD);
  float*          priorm = cf(BATCH * ZD);
  float*          priors = cf(BATCH * ZD);
  unsigned short* zbf   = cu(BATCH * ZD);
  unsigned short* phiz  = cu(BATCH * HD);
  unsigned short* dech1 = cu(BATCH * HD);
  unsigned short* dech  = cu(BATCH * HD);
  float*          gates = cf((size_t)BATCH * 2048);

  // ---- launch helpers ----
  auto cvt = [&](unsigned short* dst, const float* src, int sr, int sc, int dr, int dc) {
    int n = dr * dc;
    cvt_pad_kernel<<<(n + 255) / 256, 256, 0, stream>>>(dst, src, sr, sc, dr, dc);
  };
  auto gemm = [&](int act, const unsigned short* A, const unsigned short* W,
                  const float* bias, float* outF, int ldo, unsigned short* outH,
                  int M, int N, int K, int Nout) {
    dim3 grid((N + 127) / 128, (M + 63) / 64), block(256);
    switch (act) {
      case 0: gemm_bf16_wmma<0><<<grid, block, 0, stream>>>(A, W, bias, outF, ldo, outH, M, N, K, Nout); break;
      case 1: gemm_bf16_wmma<1><<<grid, block, 0, stream>>>(A, W, bias, outF, ldo, outH, M, N, K, Nout); break;
      case 2: gemm_bf16_wmma<2><<<grid, block, 0, stream>>>(A, W, bias, outF, ldo, outH, M, N, K, Nout); break;
      case 3: gemm_bf16_wmma<3><<<grid, block, 0, stream>>>(A, W, bias, outF, ldo, outH, M, N, K, Nout); break;
      default: gemm_bf16_wmma<4><<<grid, block, 0, stream>>>(A, W, bias, outF, ldo, outH, M, N, K, Nout); break;
    }
  };

  // ---- 1) convert / pack weights to bf16 (stay L2-resident afterwards) ----
  cvt(w_phix1, P[PHI_X_W1], 512, XD, 512, XDP);
  cvt(w_phix2, P[PHI_X_W2], 512, 512, 512, 512);
  cvt(w_phiz,  P[PHI_Z_W],  512, 128, 512, 128);
  cvt(w_enc1,  P[ENC_W1],   512, 1024, 512, 1024);
  cvt(w_enc2,  P[ENC_W2],   512, 512, 512, 512);
  cvt(w_encm,  P[ENC_MEAN_W], 128, 512, 128, 512);
  cvt(w_encs,  P[ENC_STD_W],  128, 512, 128, 512);
  cvt(w_prior, P[PRIOR_W],  512, 512, 512, 512);
  cvt(w_priorm, P[PRIOR_MEAN_W], 128, 512, 128, 512);
  cvt(w_priors, P[PRIOR_STD_W],  128, 512, 128, 512);
  cvt(w_dec1,  P[DEC_W1],   512, 1024, 512, 1024);
  cvt(w_dec2,  P[DEC_W2],   512, 512, 512, 512);
  cvt(w_decm,  P[DEC_MEAN_W], XD, 512, 96, 512);   // N padded 88 -> 96
  cvt(w_decs,  P[DEC_STD_W],  XD, 512, 96, 512);
  {
    int n = 2048 * 1536;
    lstm_pack_kernel<<<(n + 255) / 256, 256, 0, stream>>>(w_lstm, P[LSTM_W_IH], P[LSTM_W_HH]);
  }
  init_state_kernel<<<(BATCH * HD + 255) / 256, 256, 0, stream>>>(h_buf, c_buf, out);

  // ---- 2) time-parallel phi_x precompute (chunks of 16 steps) ----
  for (int t0 = 0; t0 < T_STEPS; t0 += 16) {
    int cm = 16 * BATCH;  // 8192 rows
    cvt(xpad, x + (size_t)t0 * BATCH * XD, cm, XD, cm, XDP);
    gemm(1, xpad, w_phix1, P[PHI_X_B1], nullptr, 0, h1chunk, cm, 512, XDP, 512);
    gemm(1, h1chunk, w_phix2, P[PHI_X_B2], nullptr, 0,
         phix_all + (size_t)t0 * BATCH * HD, cm, 512, 512, 512);
  }

  // ---- 3) sequential recurrence ----
  const int catN = (BATCH * 1024 + 255) / 256;
  for (int t = 0; t < T_STEPS; ++t) {
    const unsigned short* phx = phix_all + (size_t)t * BATCH * HD;
    const float* epsT  = eps + (size_t)t * BATCH * ZD;
    const float* maskT = mask + (size_t)t * BATCH;
    const float* xT    = x + (size_t)t * BATCH * XD;
    float* emT = encM + (size_t)t * BATCH * ZD;
    float* esT = encS + (size_t)t * BATCH * ZD;
    float* dmT = decM + (size_t)t * BATCH * XD;
    float* dsT = decS + (size_t)t * BATCH * XD;

    cat_start_kernel<<<catN, 256, 0, stream>>>(catA, catC, phx, h_buf);

    gemm(1, catA,  w_enc1,  P[ENC_B1],  nullptr, 0, ench1, BATCH, 512, 1024, 512);
    gemm(1, ench1, w_enc2,  P[ENC_B2],  nullptr, 0, ench,  BATCH, 512, 512, 512);
    gemm(1, h_buf, w_prior, P[PRIOR_B], nullptr, 0, priorh, BATCH, 512, 512, 512);

    gemm(0, ench,   w_encm,   P[ENC_MEAN_B],   emT,    ZD, nullptr, BATCH, 128, 512, 128);
    gemm(3, ench,   w_encs,   P[ENC_STD_B],    esT,    ZD, nullptr, BATCH, 128, 512, 128);
    gemm(0, priorh, w_priorm, P[PRIOR_MEAN_B], priorm, ZD, nullptr, BATCH, 128, 512, 128);
    gemm(4, priorh, w_priors, P[PRIOR_STD_B],  priors, ZD, nullptr, BATCH, 128, 512, 128);

    z_kernel<<<(BATCH * ZD + 255) / 256, 256, 0, stream>>>(zbf, epsT, emT, esT);
    kld_kernel<<<BATCH, 128, 0, stream>>>(kld, emT, esT, priorm, priors, maskT);

    gemm(1, zbf, w_phiz, P[PHI_Z_B], nullptr, 0, phiz, BATCH, 512, 128, 512);
    cat_mid_kernel<<<catN, 256, 0, stream>>>(catB, catC, phiz, h_buf);

    gemm(1, catB,  w_dec1, P[DEC_B1], nullptr, 0, dech1, BATCH, 512, 1024, 512);
    gemm(1, dech1, w_dec2, P[DEC_B2], nullptr, 0, dech,  BATCH, 512, 512, 512);
    gemm(2, dech, w_decm, P[DEC_MEAN_B], dmT, XD, nullptr, BATCH, 96, 512, XD);
    gemm(3, dech, w_decs, P[DEC_STD_B],  dsT, XD, nullptr, BATCH, 96, 512, XD);

    nll_kernel<<<BATCH, 128, 0, stream>>>(nll, xT, dmT, maskT);

    gemm(0, catC, w_lstm, nullptr, gates, 2048, nullptr, BATCH, 2048, 1536, 2048);
    lstm_kernel<<<(BATCH * HD + 255) / 256, 256, 0, stream>>>(h_buf, c_buf, gates);
  }
}
